// TransformerEncoder_59442347377604
// MI455X (gfx1250) — compile-verified, matrix operand order checked by reading, not run on hardware
//
#include <hip/hip_runtime.h>
#include <hip/hip_bf16.h>
#include <math.h>

// ---------------------------------------------------------------------------
// Model constants (from the reference)
// ---------------------------------------------------------------------------
#define V_SZ   32000
#define PAD_ID 0
#define D_SZ   1024
#define L_SZ   6
#define H_SZ   16
#define FF_SZ  4096
#define NB_SZ  32
#define MD_SZ  128
#define B_SZ   2
#define S_SZ   1024
#define DH_SZ  64          // D/H
#define M_TOK  (B_SZ * S_SZ)   // 2048 token rows

typedef __bf16 bf16;
typedef __bf16 v8bf  __attribute__((ext_vector_type(8)));
typedef __bf16 v16bf __attribute__((ext_vector_type(16)));
typedef float  v8f   __attribute__((ext_vector_type(8)));

// CDNA5 16-bit A/B fragment (16 x K=32 per lane) from two 16B chunks:
// lanes 0-15 carry K=[0..7]+[16..23], lanes 16-31 carry K=[8..15]+[24..31].
__device__ inline v16bf make_frag(v8bf lo, v8bf hi) {
    v16bf f;
#pragma unroll
    for (int i = 0; i < 8; ++i) { f[i] = lo[i]; f[i + 8] = hi[i]; }
    return f;
}

__device__ inline v8f wmma_bf16(v16bf a, v16bf b, v8f c) {
    return __builtin_amdgcn_wmma_f32_16x16x32_bf16(
        false, a, false, b, (short)0, c, false, false);
}

// Async global->LDS copy of 16 bytes per lane (ASYNCcnt-tracked, bypasses
// VGPRs).  GV mode: 64-bit per-lane global address, per-lane 32-bit LDS
// byte address in VDST.  Low 32 bits of a generic __shared__ pointer are the
// AS3 offset, which is what the instruction wants.
__device__ inline void async_copy_b128(void* lds_ptr, const void* gptr) {
    unsigned lds32 = (unsigned)(unsigned long long)lds_ptr;
    asm volatile("global_load_async_to_lds_b128 %0, %1, off"
                 :: "v"(lds32), "v"(gptr) : "memory");
}
__device__ inline void wait_async0() {
    asm volatile("s_wait_asynccnt 0x0" ::: "memory");
}

// ---------------------------------------------------------------------------
// 1) fp32 -> bf16 conversion (weights)
// ---------------------------------------------------------------------------
__global__ void f32_to_bf16_kernel(const float* __restrict__ src,
                                   bf16* __restrict__ dst, int n) {
    int i = blockIdx.x * blockDim.x + threadIdx.x;
    if (i < n) dst[i] = (bf16)src[i];
}

// ---------------------------------------------------------------------------
// 2) embedding gather * sqrt(D)  (x stays fp32: residual stream)
// ---------------------------------------------------------------------------
__global__ void embed_kernel(const int* __restrict__ ids,
                             const float* __restrict__ embed,
                             float* __restrict__ x) {
    int i = blockIdx.x * blockDim.x + threadIdx.x;   // 0 .. B*S*D-1
    int row = i >> 10;           // /D
    int col = i & (D_SZ - 1);
    int id  = ids[row];
    x[i] = embed[(size_t)id * D_SZ + col] * 32.0f;   // sqrt(1024)=32
}

// ---------------------------------------------------------------------------
// 3) T5 relative bias table: biasT[(q-k)+S-1][h]
// ---------------------------------------------------------------------------
__global__ void bias_table_kernel(const float* __restrict__ rel_emb,
                                  float* __restrict__ biasT) {
    int g = blockIdx.x * blockDim.x + threadIdx.x;
    if (g >= (2 * S_SZ - 1) * H_SZ) return;
    int d = g >> 4;          // 0..2046
    int h = g & 15;
    int n = d - (S_SZ - 1);  // n = q - k
    const int nb = NB_SZ / 2;        // 16
    const int max_exact = nb / 2;    // 8
    int ret = (n < 0) ? nb : 0;
    int na  = n < 0 ? -n : n;
    int bucket;
    if (na < max_exact) {
        bucket = ret + na;
    } else {
        int vl = max_exact + (int)(logf((float)na / (float)max_exact + 1e-6f) /
                                   logf((float)MD_SZ / (float)max_exact) *
                                   (float)(nb - max_exact));
        vl = vl < (nb - 1) ? vl : (nb - 1);
        bucket = ret + vl;
    }
    biasT[g] = rel_emb[bucket * H_SZ + h];
}

// ---------------------------------------------------------------------------
// 4) RMSNorm: one 256-thread block per row of D=1024.
//    Writes bf16 (GEMM input) or fp32 (final output).
// ---------------------------------------------------------------------------
__global__ void rmsnorm_kernel(const float* __restrict__ x,
                               const float* __restrict__ w,
                               float* __restrict__ of,
                               bf16* __restrict__ ob) {
    __shared__ float red[256];
    int row = blockIdx.x;
    int tid = threadIdx.x;
    const float* xp = x + (size_t)row * D_SZ;
    float vals[4];
    float ss = 0.f;
#pragma unroll
    for (int i = 0; i < 4; ++i) {
        vals[i] = xp[tid + i * 256];
        ss += vals[i] * vals[i];
    }
    red[tid] = ss;
    __syncthreads();
    for (int s = 128; s > 0; s >>= 1) {
        if (tid < s) red[tid] += red[tid + s];
        __syncthreads();
    }
    float r = rsqrtf(red[0] * (1.0f / D_SZ) + 1e-6f);
#pragma unroll
    for (int i = 0; i < 4; ++i) {
        int c = tid + i * 256;
        float v = vals[i] * r * w[c];
        if (ob) ob[(size_t)row * D_SZ + c] = (bf16)v;
        else    of[(size_t)row * D_SZ + c] = v;
    }
}

// ---------------------------------------------------------------------------
// 5) WMMA GEMM: out[M,N] = A[M,K](bf16) * W[N,K]^T(bf16)
//    128x128x32 block tile, 8 waves (2x4), 4x2 accumulators per wave.
//    Double-buffered LDS staged with GLOBAL_LOAD_ASYNC_TO_LDS_B128.
//    mode 1: store bf16
//    mode 2: +bias, exact GELU, store bf16
//    mode 3: +bias(null ok), += residual, store fp32
// ---------------------------------------------------------------------------
__global__ void gemm_bf16_kernel(const bf16* __restrict__ A,
                                 const bf16* __restrict__ W,
                                 const float* __restrict__ bias,
                                 float* __restrict__ outF,
                                 bf16* __restrict__ outB,
                                 int M, int N, int K, int mode) {
    __shared__ bf16 As[2][128][40];   // padded stride, rows 16B-aligned
    __shared__ bf16 Bs[2][128][40];

    const int tid  = threadIdx.x;
    const int wid  = tid >> 5;
    const int lane = tid & 31;
    const int l15  = lane & 15;
    const int lhi  = lane >> 4;
    const int wm   = wid >> 2;      // 0..1  -> 64 M rows
    const int wn   = wid & 3;       // 0..3  -> 32 N cols

    const int rowA = blockIdx.y * 128;
    const int rowB = blockIdx.x * 128;

    const int r  = tid >> 1;        // tile row this thread stages
    const int ch = (tid & 1) * 16;  // k half (16 bf16 = 32B = 2 x b128)

    v8f acc[4][2] = {};

    // stage tile (kk) of A and B into LDS buffer `buf` (async, 4 x b128/lane)
    auto stage = [&](int buf, int kk) {
        const bf16* ga = A + (size_t)(rowA + r) * K + kk + ch;
        const bf16* gb = W + (size_t)(rowB + r) * K + kk + ch;
        async_copy_b128(&As[buf][r][ch],     ga);
        async_copy_b128(&As[buf][r][ch + 8], ga + 8);
        async_copy_b128(&Bs[buf][r][ch],     gb);
        async_copy_b128(&Bs[buf][r][ch + 8], gb + 8);
    };

    stage(0, 0);
    wait_async0();
    __syncthreads();

    int buf = 0;
    for (int kk = 0; kk < K; kk += 32) {
        if (kk + 32 < K) stage(buf ^ 1, kk + 32);

        v16bf aF[4], bF[2];
#pragma unroll
        for (int i = 0; i < 4; ++i) {
            int mr = wm * 64 + i * 16 + l15;
            aF[i] = make_frag(*(const v8bf*)&As[buf][mr][8 * lhi],
                              *(const v8bf*)&As[buf][mr][8 * lhi + 16]);
        }
#pragma unroll
        for (int j = 0; j < 2; ++j) {
            int nr = wn * 32 + j * 16 + l15;
            bF[j] = make_frag(*(const v8bf*)&Bs[buf][nr][8 * lhi],
                              *(const v8bf*)&Bs[buf][nr][8 * lhi + 16]);
        }
#pragma unroll
        for (int i = 0; i < 4; ++i)
#pragma unroll
            for (int j = 0; j < 2; ++j)
                acc[i][j] = wmma_bf16(aF[i], bF[j], acc[i][j]);

        wait_async0();        // next buffer landed
        __syncthreads();      // everyone done reading current buffer
        buf ^= 1;
    }

    // ---- epilogue ----
#pragma unroll
    for (int i = 0; i < 4; ++i) {
#pragma unroll
        for (int j = 0; j < 2; ++j) {
            int gn = rowB + wn * 32 + j * 16 + l15;
#pragma unroll
            for (int rr = 0; rr < 8; ++rr) {
                int gm = rowA + wm * 64 + i * 16 + rr + 8 * lhi;
                size_t idx = (size_t)gm * N + gn;
                float v = acc[i][j][rr];
                if (mode == 1) {
                    outB[idx] = (bf16)v;
                } else if (mode == 2) {
                    v += bias[gn];
                    v = 0.5f * v * (1.0f + erff(v * 0.70710678118f));
                    outB[idx] = (bf16)v;
                } else { // mode 3
                    if (bias) v += bias[gn];
                    outF[idx] = outF[idx] + v;
                }
            }
        }
    }
}

// ---------------------------------------------------------------------------
// 6) Fused attention: per block = (b, h, 32 q rows).
//    scores = QK^T*scale + bias(q-k), key-padding mask, fp32 softmax in LDS,
//    P(bf16) x V via WMMA with an LDS-transposed V tile.  Output bf16.
// ---------------------------------------------------------------------------
#define SCS 1032   // score row stride (floats; 16B multiple as bf16 too)

__global__ void attention_kernel(const bf16* __restrict__ q,
                                 const bf16* __restrict__ k,
                                 const bf16* __restrict__ v,
                                 const float* __restrict__ biasT,
                                 const int* __restrict__ ids,
                                 bf16* __restrict__ out) {
    extern __shared__ char smem[];
    float* sc     = (float*)smem;                                   // 32*SCS f32
    bf16*  P      = (bf16*)(smem + 32 * SCS * sizeof(float));       // 32*SCS bf16
    bf16*  vT     = (bf16*)((char*)P + 32 * SCS * sizeof(bf16));    // 64*40 bf16
    float* red    = (float*)((char*)vT + 64 * 40 * sizeof(bf16));   // 256 f32
    float* rowmax = red + 256;                                      // 32
    float* rowsum = rowmax + 32;                                    // 32

    const int tid  = threadIdx.x;
    const int wid  = tid >> 5;
    const int lane = tid & 31;
    const int l15  = lane & 15;
    const int lhi  = lane >> 4;

    const int qbase = blockIdx.x * 32;
    const int b     = blockIdx.y >> 4;
    const int h     = blockIdx.y & 15;
    const float scale = 0.125f;              // DH^-0.5
    const float NEG   = -3.4028235e38f;

    // ---------------- score phase: waves = 2 qsub x 4 k-quarters ----------
    {
        const int qsub    = wid & 1;
        const int quarter = wid >> 1;
        const int kbase0  = quarter * 256;

        v16bf aF[2];
#pragma unroll
        for (int t = 0; t < 2; ++t) {
            int m = qbase + qsub * 16 + l15;
            const bf16* qp = q + (size_t)(b * S_SZ + m) * D_SZ +
                             h * DH_SZ + t * 32 + 8 * lhi;
            aF[t] = make_frag(*(const v8bf*)qp, *(const v8bf*)(qp + 16));
        }
        for (int tile = 0; tile < 16; ++tile) {
            int kc = kbase0 + tile * 16 + l15;
            v8f accS = {};
#pragma unroll
            for (int t = 0; t < 2; ++t) {
                const bf16* kp = k + (size_t)(b * S_SZ + kc) * D_SZ +
                                 h * DH_SZ + t * 32 + 8 * lhi;
                v16bf bF = make_frag(*(const v8bf*)kp, *(const v8bf*)(kp + 16));
                accS = wmma_bf16(aF[t], bF, accS);
            }
#pragma unroll
            for (int rr = 0; rr < 8; ++rr) {
                int lq   = qsub * 16 + rr + 8 * lhi;
                int qpos = qbase + lq;
                int kpos = kc;
                float s = accS[rr] * scale +
                          biasT[(qpos - kpos + S_SZ - 1) * H_SZ + h];
                if (ids[b * S_SZ + kpos] == PAD_ID) s = NEG;
                sc[lq * SCS + kpos] = s;
            }
        }
    }
    __syncthreads();

    // ---------------- softmax: 8 threads per row --------------------------
    {
        const int row = tid >> 3;
        const int tpr = tid & 7;
        float m = NEG;
        for (int c = tpr; c < S_SZ; c += 8) m = fmaxf(m, sc[row * SCS + c]);
        red[tid] = m;
        __syncthreads();
        if (tpr == 0) {
            float mm = red[row * 8];
#pragma unroll
            for (int j = 1; j < 8; ++j) mm = fmaxf(mm, red[row * 8 + j]);
            rowmax[row] = mm;
        }
        __syncthreads();
        float rm = rowmax[row];
        float ssum = 0.f;
        for (int c = tpr; c < S_SZ; c += 8) {
            float e = __expf(sc[row * SCS + c] - rm);
            sc[row * SCS + c] = e;
            ssum += e;
        }
        red[tid] = ssum;
        __syncthreads();
        if (tpr == 0) {
            float s8 = red[row * 8];
#pragma unroll
            for (int j = 1; j < 8; ++j) s8 += red[row * 8 + j];
            rowsum[row] = s8;
        }
        __syncthreads();
        float inv = 1.0f / rowsum[row];
        for (int c = tpr; c < S_SZ; c += 8)
            P[row * SCS + c] = (bf16)(sc[row * SCS + c] * inv);
    }
    __syncthreads();

    // ---------------- out = P x V: waves = 2 qsub x 4 dh-subtiles ---------
    {
        const int qsub  = wid & 1;
        const int dhsub = wid >> 1;
        v8f accO = {};
        for (int kk = 0; kk < S_SZ; kk += 32) {
            int vr = tid >> 3;               // 0..31 key row
            int vc = (tid & 7) * 8;          // dh octet
            const bf16* vp = v + (size_t)(b * S_SZ + kk + vr) * D_SZ +
                             h * DH_SZ + vc;
            v8bf vv = *(const v8bf*)vp;
#pragma unroll
            for (int e = 0; e < 8; ++e) vT[(vc + e) * 40 + vr] = vv[e];
            __syncthreads();

            int lq = qsub * 16 + l15;
            v16bf aF = make_frag(*(const v8bf*)&P[lq * SCS + kk + 8 * lhi],
                                 *(const v8bf*)&P[lq * SCS + kk + 8 * lhi + 16]);
            int dh = dhsub * 16 + l15;
            v16bf bF = make_frag(*(const v8bf*)&vT[dh * 40 + 8 * lhi],
                                 *(const v8bf*)&vT[dh * 40 + 8 * lhi + 16]);
            accO = wmma_bf16(aF, bF, accO);
            __syncthreads();
        }
#pragma unroll
        for (int rr = 0; rr < 8; ++rr) {
            int m   = qbase + qsub * 16 + rr + 8 * lhi;
            int col = h * DH_SZ + dhsub * 16 + l15;
            out[(size_t)(b * S_SZ + m) * D_SZ + col] = (bf16)accO[rr];
        }
    }
}

// ---------------------------------------------------------------------------
// 7) key-padding mask -> d_out tail (0/1 floats)
// ---------------------------------------------------------------------------
__global__ void mask_kernel(const int* __restrict__ ids, float* __restrict__ o) {
    int i = blockIdx.x * blockDim.x + threadIdx.x;
    if (i < B_SZ * S_SZ) o[i] = (ids[i] == PAD_ID) ? 1.0f : 0.0f;
}

// ---------------------------------------------------------------------------
// Host: workspace layout + launch sequence
// ---------------------------------------------------------------------------
static inline size_t align256(size_t x) { return (x + 255) & ~(size_t)255; }

extern "C" void kernel_launch(void* const* d_in, const int* in_sizes, int n_in,
                              void* d_out, int out_size, void* d_ws, size_t ws_size,
                              hipStream_t stream) {
    (void)in_sizes; (void)n_in; (void)out_size; (void)ws_size;

    const int*   ids     = (const int*)  d_in[0];
    const float* embed   = (const float*)d_in[1];
    const float* rel_emb = (const float*)d_in[2];
    const float* qW      = (const float*)d_in[3];
    const float* kW      = (const float*)d_in[4];
    const float* vW      = (const float*)d_in[5];
    const float* oW      = (const float*)d_in[6];
    const float* fc1W    = (const float*)d_in[7];
    const float* fc1b    = (const float*)d_in[8];
    const float* fc2W    = (const float*)d_in[9];
    const float* fc2b    = (const float*)d_in[10];
    const float* n1w     = (const float*)d_in[11];
    const float* n2w     = (const float*)d_in[12];
    const float* fnw     = (const float*)d_in[13];

    const size_t nProj = (size_t)L_SZ * D_SZ * D_SZ;    // 6M elems
    const size_t nFF   = (size_t)L_SZ * FF_SZ * D_SZ;   // 24M elems
    char* ws = (char*)d_ws;
    size_t off = 0;
    bf16* qWb   = (bf16*)(ws + off); off = align256(off + nProj * 2);
    bf16* kWb   = (bf16*)(ws + off); off = align256(off + nProj * 2);
    bf16* vWb   = (bf16*)(ws + off); off = align256(off + nProj * 2);
    bf16* oWb   = (bf16*)(ws + off); off = align256(off + nProj * 2);
    bf16* fc1Wb = (bf16*)(ws + off); off = align256(off + nFF * 2);
    bf16* fc2Wb = (bf16*)(ws + off); off = align256(off + nFF * 2);
    float* x    = (float*)(ws + off); off = align256(off + (size_t)M_TOK * D_SZ * 4);
    bf16* hbuf  = (bf16*)(ws + off); off = align256(off + (size_t)M_TOK * D_SZ * 2);
    bf16* attnB = (bf16*)(ws + off); off = align256(off + (size_t)M_TOK * D_SZ * 2);
    bf16* h2    = (bf16*)(ws + off); off = align256(off + (size_t)M_TOK * FF_SZ * 2);
    bf16* qA    = (bf16*)(ws + off); off = align256(off + (size_t)M_TOK * D_SZ * 2);
    bf16* kA    = (bf16*)(ws + off); off = align256(off + (size_t)M_TOK * D_SZ * 2);
    bf16* vA    = (bf16*)(ws + off); off = align256(off + (size_t)M_TOK * D_SZ * 2);
    float* biasT = (float*)(ws + off); off = align256(off + (size_t)(2 * S_SZ - 1) * H_SZ * 4);

    // ---- 1. convert weights to bf16 ----
    {
        int np = (int)nProj, nf = (int)nFF;
        f32_to_bf16_kernel<<<(np + 255) / 256, 256, 0, stream>>>(qW,   qWb,   np);
        f32_to_bf16_kernel<<<(np + 255) / 256, 256, 0, stream>>>(kW,   kWb,   np);
        f32_to_bf16_kernel<<<(np + 255) / 256, 256, 0, stream>>>(vW,   vWb,   np);
        f32_to_bf16_kernel<<<(np + 255) / 256, 256, 0, stream>>>(oW,   oWb,   np);
        f32_to_bf16_kernel<<<(nf + 255) / 256, 256, 0, stream>>>(fc1W, fc1Wb, nf);
        f32_to_bf16_kernel<<<(nf + 255) / 256, 256, 0, stream>>>(fc2W, fc2Wb, nf);
    }

    // ---- 2. embed + 3. bias table ----
    embed_kernel<<<(M_TOK * D_SZ) / 256, 256, 0, stream>>>(ids, embed, x);
    bias_table_kernel<<<((2 * S_SZ - 1) * H_SZ + 255) / 256, 256, 0, stream>>>(rel_emb, biasT);

    const dim3 gProj(D_SZ / 128, M_TOK / 128);   // (8,16)
    const dim3 gFF1(FF_SZ / 128, M_TOK / 128);   // (32,16)
    const dim3 gAtt(S_SZ / 32, B_SZ * H_SZ);     // (32,32)
    const size_t attLds = 32 * SCS * sizeof(float) + 32 * SCS * sizeof(bf16) +
                          64 * 40 * sizeof(bf16) + (256 + 64) * sizeof(float);

    // ---- 4. layers ----
    for (int i = 0; i < L_SZ; ++i) {
        const bf16* qWi = qWb + (size_t)i * D_SZ * D_SZ;
        const bf16* kWi = kWb + (size_t)i * D_SZ * D_SZ;
        const bf16* vWi = vWb + (size_t)i * D_SZ * D_SZ;
        const bf16* oWi = oWb + (size_t)i * D_SZ * D_SZ;
        const bf16* f1i = fc1Wb + (size_t)i * FF_SZ * D_SZ;
        const bf16* f2i = fc2Wb + (size_t)i * FF_SZ * D_SZ;

        rmsnorm_kernel<<<M_TOK, 256, 0, stream>>>(x, n1w + i * D_SZ, nullptr, hbuf);

        gemm_bf16_kernel<<<gProj, 256, 0, stream>>>(hbuf, qWi, nullptr, nullptr, qA,
                                                    M_TOK, D_SZ, D_SZ, 1);
        gemm_bf16_kernel<<<gProj, 256, 0, stream>>>(hbuf, kWi, nullptr, nullptr, kA,
                                                    M_TOK, D_SZ, D_SZ, 1);
        gemm_bf16_kernel<<<gProj, 256, 0, stream>>>(hbuf, vWi, nullptr, nullptr, vA,
                                                    M_TOK, D_SZ, D_SZ, 1);

        attention_kernel<<<gAtt, 256, attLds, stream>>>(qA, kA, vA, biasT, ids, attnB);

        // x += attn @ oW^T
        gemm_bf16_kernel<<<gProj, 256, 0, stream>>>(attnB, oWi, nullptr, x, nullptr,
                                                    M_TOK, D_SZ, D_SZ, 3);

        rmsnorm_kernel<<<M_TOK, 256, 0, stream>>>(x, n2w + i * D_SZ, nullptr, hbuf);

        // h2 = gelu(h @ fc1^T + b1)   (bf16)
        gemm_bf16_kernel<<<gFF1, 256, 0, stream>>>(hbuf, f1i, fc1b + i * FF_SZ,
                                                   nullptr, h2, M_TOK, FF_SZ, D_SZ, 2);
        // x += h2 @ fc2^T + b2
        gemm_bf16_kernel<<<gProj, 256, 0, stream>>>(h2, f2i, fc2b + i * D_SZ,
                                                    x, nullptr, M_TOK, D_SZ, FF_SZ, 3);
    }

    // ---- 5. final norm -> d_out, 6. mask tail ----
    float* outF = (float*)d_out;
    rmsnorm_kernel<<<M_TOK, 256, 0, stream>>>(x, fnw, outF, nullptr);
    mask_kernel<<<(B_SZ * S_SZ + 255) / 256, 256, 0, stream>>>(ids, outF + (size_t)M_TOK * D_SZ);
}
